// KerasSeq2Seq_8022998909631
// MI455X (gfx1250) — matrix operand order repaired and checked by reading, not compiled, not run on hardware
//
#include <hip/hip_runtime.h>
#include <cstdint>
#include <cstddef>

// ---------------------------------------------------------------------------
// CDNA5 (gfx1250) persistent fused seq2seq-LSTM, v3.
//   v2 -> v3: cap kc-loop unrolling (#pragma unroll 2). v2's full unroll made
//   the scheduler hoist 64 weight-fragment loads, spilling them to scratch and
//   serializing every WMMA on a scratch reload. Rolled loop keeps live set
//   ~140 VGPRs: acc 32 + c 16 + A 8 + 2x4 B frags 64 + addressing.
// ---------------------------------------------------------------------------

typedef __attribute__((ext_vector_type(16))) __bf16        v16bf;
typedef __attribute__((ext_vector_type(8)))  float         v8f;
typedef __attribute__((ext_vector_type(4)))  unsigned int  u32x4;

#define B_    1024
#define T_    64
#define SEG_  32
#define H_    512
#define F_    121
#define FP_   128          // F padded to multiple of 32 (zeros beyond 121)
#define G4_   2048         // 4*H gate width
#define MT_   16           // batch rows per workgroup (one WMMA M tile)
#define HS_   520          // LDS row stride in halves (512 + 8 pad)
#define XS_   136          // LDS row stride for x tile (128 + 8 pad)
#define NTHR_ 1024         // 32 waves

// ---- bf16 helpers (round-to-nearest-even) ----------------------------------
static __device__ __forceinline__ unsigned short f2bf(float f) {
    unsigned int u = __float_as_uint(f);
    unsigned int r = (u + 0x7FFFu + ((u >> 16) & 1u)) >> 16;
    return (unsigned short)r;
}
static __device__ __forceinline__ float bf2f(unsigned short h) {
    return __uint_as_float(((unsigned int)h) << 16);
}
static __device__ __forceinline__ float sigm(float x) {
    return 1.0f / (1.0f + __expf(-x));
}

// ---------------------------------------------------------------------------
// Prep 1: x [B,T,F] f32 -> bf16 [B,T,FP] (zero padded K)
// ---------------------------------------------------------------------------
__global__ void prep_x_kernel(const float* __restrict__ x,
                              unsigned short* __restrict__ xbf) {
    size_t i = (size_t)blockIdx.x * blockDim.x + threadIdx.x;
    size_t n = (size_t)B_ * T_ * FP_;
    if (i >= n) return;
    int c = (int)(i % FP_);
    size_t bt = i / FP_;
    float v = (c < F_) ? x[bt * F_ + c] : 0.0f;
    xbf[i] = f2bf(v);
}

// ---------------------------------------------------------------------------
// Prep 2: weight [K,2048] f32 -> bf16 WMMA B-fragment blocks.
//   block (kc, nt): K rows kc*32..+31, cols nt*16..+15.
//   lane l<16 -> col l,   K 0..7,16..23 ; lane l>=16 -> col l-16, K 8..15,24..31
//   16 contiguous halves per lane -> one coalesced 32B load in the hot loop.
// ---------------------------------------------------------------------------
__global__ void prep_w_kernel(const float* __restrict__ w,
                              unsigned short* __restrict__ dst,
                              int K, int KC) {
    size_t i = (size_t)blockIdx.x * blockDim.x + threadIdx.x;
    size_t n = (size_t)KC * 128 * 512;
    if (i >= n) return;
    int e    = (int)(i & 15);
    int lane = (int)((i >> 4) & 31);
    size_t blk = i >> 9;
    int nt = (int)(blk & 127);
    int kc = (int)(blk >> 7);
    int k   = kc * 32 + ((e < 8) ? e : 8 + e) + ((lane < 16) ? 0 : 8);
    int col = nt * 16 + (lane & 15);
    float v = (k < K) ? w[(size_t)k * G4_ + col] : 0.0f;
    dst[i] = f2bf(v);
}

// ---- WMMA fragment loads ----------------------------------------------------
union Frag32B { v16bf v; u32x4 q[2]; };

// z += A(lds) x B(weights) over kcN K-chunks; 4 tiles/wave: nt = g*32 + wave.
// A (16x32 bf16) ISA layout: lane l<16: row l, K c..c+7 & c+16..c+23;
//                            lane l>=16: row l-16, +8.
static __device__ __forceinline__ void gemm(const unsigned short* abase, int astride,
                                            const unsigned short* __restrict__ wf,
                                            int kcN, int wave, int lane, v8f acc[4]) {
    // per-lane invariant addressing
    const unsigned short* ap =
        abase + (lane & 15) * astride + ((lane >> 4) << 3);
    const unsigned short* bp = wf + ((size_t)wave << 9) + (lane << 4);

#pragma unroll 2
    for (int kc = 0; kc < kcN; ++kc) {
        Frag32B a;
        a.q[0] = *(const u32x4*)(ap);        // ds_load_b128 (K .. K+7)
        a.q[1] = *(const u32x4*)(ap + 16);   // ds_load_b128 (K+16..K+23)
        ap += 32;
#pragma unroll
        for (int g = 0; g < 4; ++g) {
            v16bf b = *(const v16bf*)(bp + ((size_t)g << 14)); // +g*32 tiles
            acc[g] = __builtin_amdgcn_wmma_f32_16x16x32_bf16(
                false, a.v, false, b, (short)0, acc[g], false, false);
        }
        bp += (size_t)128 << 9;              // next K-chunk (128 tiles * 512 halves)
    }
}

static __device__ __forceinline__ void init_acc(v8f acc[4], const float bl[4]) {
#pragma unroll
    for (int g = 0; g < 4; ++g)
#pragma unroll
        for (int r = 0; r < 8; ++r)
            acc[g][r] = bl[g];
}

// gates (i,f,g,o) -> c,h ; c stays in registers, h (bf16) goes to LDS.
// C/D layout: VGPR r, lane l -> row r + (l>=16)*8, col = l&15 within tile.
static __device__ __forceinline__ void lstm_update(v8f z[4], v8f& c,
                                                   unsigned short* hbase,
                                                   int stride, int wave, int lane) {
    int col  = wave * 16 + (lane & 15);
    int rowb = ((lane >> 4) & 1) << 3;
#pragma unroll
    for (int r = 0; r < 8; ++r) {
        float iv = sigm(z[0][r]);
        float fv = sigm(z[1][r]);
        float gv = tanhf(z[2][r]);
        float ov = sigm(z[3][r]);
        float cv = fv * c[r] + iv * gv;
        c[r] = cv;
        hbase[(rowb + r) * stride + col] = f2bf(ov * tanhf(cv));
    }
}

// ---------------------------------------------------------------------------
// Main persistent kernel: whole model for one 16-row batch tile.
// ---------------------------------------------------------------------------
__global__ __launch_bounds__(NTHR_, 1)
void seq2seq_lstm_kernel(const unsigned short* __restrict__ xbf,
                         const unsigned short* __restrict__ wE0f,
                         const unsigned short* __restrict__ uE0f,
                         const unsigned short* __restrict__ wE1f,
                         const unsigned short* __restrict__ uE1f,
                         const unsigned short* __restrict__ uD0f,
                         const unsigned short* __restrict__ wD1f,
                         const unsigned short* __restrict__ uD1f,
                         const float* __restrict__ eb0, const float* __restrict__ eb1,
                         const float* __restrict__ db0, const float* __restrict__ db1,
                         const float* __restrict__ denseW,
                         const float* __restrict__ denseb,
                         float* __restrict__ out) {
    __shared__ unsigned short hA[MT_ * HS_];   // layer-0 h (enc0 / dec0)
    __shared__ unsigned short hB[MT_ * HS_];   // layer-1 h (enc1 / dec1)
    __shared__ unsigned short xls[MT_ * XS_];  // x_t tile, bf16, K padded
    __shared__ float          red[MT_];        // dense reduction

    const int tid  = threadIdx.x;
    const int lane = tid & 31;
    const int wave = tid >> 5;                 // 0..31, owns hidden cols wave*16..+15
    const int b0   = blockIdx.x * MT_;

    for (int i = tid; i < MT_ * HS_; i += NTHR_) { hA[i] = 0; hB[i] = 0; }

    // per-wave bias fragments (value depends only on output column)
    float blE0[4], blE1[4], blD0[4], blD1[4];
#pragma unroll
    for (int g = 0; g < 4; ++g) {
        int idx = g * 512 + wave * 16 + (lane & 15);
        blE0[g] = eb0[idx];
        blE1[g] = eb1[idx];
        blD0[g] = db0[idx];
        blD1[g] = db1[idx];
    }

    v8f c0, c1;
#pragma unroll
    for (int r = 0; r < 8; ++r) { c0[r] = 0.0f; c1[r] = 0.0f; }

    __syncthreads();

    // ================= encoder: both layers fused per step ==================
    for (int t = 0; t < T_; ++t) {
        { // stage x_t tile (16 rows x 128 halves): 1 uint (2 halves) per thread
            int r  = tid >> 6;
            int c2 = (tid & 63) << 1;
            const unsigned short* src =
                xbf + (((size_t)(b0 + r) * T_ + t) * FP_ + c2);
            *(unsigned int*)(&xls[r * XS_ + c2]) = *(const unsigned int*)src;
        }
        __syncthreads();

        v8f acc[4];
        // enc layer 0: z = x@W0 + h0@U0 + b0
        init_acc(acc, blE0);
        gemm(xls, XS_, wE0f, FP_ / 32, wave, lane, acc);
        gemm(hA,  HS_, uE0f, H_  / 32, wave, lane, acc);
        __syncthreads();                          // all hA reads done
        lstm_update(acc, c0, hA, HS_, wave, lane);
        __syncthreads();                          // new h0 visible

        // enc layer 1: z = h0@W1 + h1@U1 + b1   (input is fresh h0 in hA)
        init_acc(acc, blE1);
        gemm(hA, HS_, wE1f, H_ / 32, wave, lane, acc);
        gemm(hB, HS_, uE1f, H_ / 32, wave, lane, acc);
        __syncthreads();
        lstm_update(acc, c1, hB, HS_, wave, lane);
        __syncthreads();
    }

    // ================= decoder (dec_in == 0, so x@dW0 vanishes) =============
    const float dbias = denseb[0];
    for (int t = 0; t < SEG_; ++t) {
        v8f acc[4];
        // dec layer 0: z = h@dU0 + db0  (seeded by enc0 finals already in hA/c0)
        init_acc(acc, blD0);
        gemm(hA, HS_, uD0f, H_ / 32, wave, lane, acc);
        __syncthreads();
        lstm_update(acc, c0, hA, HS_, wave, lane);
        __syncthreads();

        // dec layer 1: z = d0@dW1 + h@dU1 + db1
        init_acc(acc, blD1);
        gemm(hA, HS_, wD1f, H_ / 32, wave, lane, acc);
        gemm(hB, HS_, uD1f, H_ / 32, wave, lane, acc);
        __syncthreads();
        lstm_update(acc, c1, hB, HS_, wave, lane);
        if (tid < MT_) red[tid] = 0.0f;
        __syncthreads();

        { // Dense(1): out[b,t] = h1 . denseW + denseb  (LDS-atomic reduce)
            int r  = tid >> 6;            // 16 rows, 64 threads per row
            int k0 = (tid & 63) << 3;     // 8 elements per thread
            float p = 0.0f;
#pragma unroll
            for (int k = 0; k < 8; ++k)
                p += bf2f(hB[r * HS_ + k0 + k]) * denseW[k0 + k];
            atomicAdd(&red[r], p);
        }
        __syncthreads();
        if (tid < MT_) out[(size_t)(b0 + tid) * SEG_ + t] = red[tid] + dbias;
        __syncthreads();
    }
}

// ---------------------------------------------------------------------------
extern "C" void kernel_launch(void* const* d_in, const int* in_sizes, int n_in,
                              void* d_out, int out_size, void* d_ws, size_t ws_size,
                              hipStream_t stream) {
    (void)in_sizes; (void)n_in; (void)out_size; (void)ws_size;

    const float* x      = (const float*)d_in[0];
    // d_in[1] dec_in == zeros (unused)
    const float* eW0    = (const float*)d_in[2];
    const float* eU0    = (const float*)d_in[3];
    const float* eb0    = (const float*)d_in[4];
    const float* eW1    = (const float*)d_in[5];
    const float* eU1    = (const float*)d_in[6];
    const float* eb1    = (const float*)d_in[7];
    // d_in[8] dW0 multiplies zeros (unused)
    const float* dU0    = (const float*)d_in[9];
    const float* db0    = (const float*)d_in[10];
    const float* dW1    = (const float*)d_in[11];
    const float* dU1    = (const float*)d_in[12];
    const float* db1    = (const float*)d_in[13];
    const float* denseW = (const float*)d_in[14];
    const float* denseb = (const float*)d_in[15];
    float* out = (float*)d_out;

    unsigned char* ws = (unsigned char*)d_ws;
    size_t off = 0;
    auto carve = [&](size_t bytes) -> unsigned short* {
        unsigned short* p = (unsigned short*)(ws + off);
        off += (bytes + 255) & ~(size_t)255;
        return p;
    };
    const size_t WSZ16 = (size_t)16 * 128 * 512 * 2;   // K=512 weight frags
    const size_t WSZ4  = (size_t)4  * 128 * 512 * 2;   // K=128 (padded F)

    unsigned short* xbf  = carve((size_t)B_ * T_ * FP_ * 2);
    unsigned short* wE0f = carve(WSZ4);
    unsigned short* uE0f = carve(WSZ16);
    unsigned short* wE1f = carve(WSZ16);
    unsigned short* uE1f = carve(WSZ16);
    unsigned short* uD0f = carve(WSZ16);
    unsigned short* wD1f = carve(WSZ16);
    unsigned short* uD1f = carve(WSZ16);

    {
        size_t n = (size_t)B_ * T_ * FP_;
        prep_x_kernel<<<(int)((n + 255) / 256), 256, 0, stream>>>(x, xbf);
    }
    {
        size_t n16 = (size_t)16 * 128 * 512;
        size_t n4  = (size_t)4  * 128 * 512;
        int b16 = (int)((n16 + 255) / 256);
        int b4  = (int)((n4  + 255) / 256);
        prep_w_kernel<<<b4,  256, 0, stream>>>(eW0, wE0f, F_, 4);
        prep_w_kernel<<<b16, 256, 0, stream>>>(eU0, uE0f, H_, 16);
        prep_w_kernel<<<b16, 256, 0, stream>>>(eW1, wE1f, H_, 16);
        prep_w_kernel<<<b16, 256, 0, stream>>>(eU1, uE1f, H_, 16);
        prep_w_kernel<<<b16, 256, 0, stream>>>(dU0, uD0f, H_, 16);
        prep_w_kernel<<<b16, 256, 0, stream>>>(dW1, wD1f, H_, 16);
        prep_w_kernel<<<b16, 256, 0, stream>>>(dU1, uD1f, H_, 16);
    }
    // 64 WGs x 1024 threads (32 waves)
    seq2seq_lstm_kernel<<<B_ / MT_, NTHR_, 0, stream>>>(
        xbf, wE0f, uE0f, wE1f, uE1f, uD0f, wD1f, uD1f,
        eb0, eb1, db0, db1, denseW, denseb, out);
}
